// TWDSGCN2_24550033064287
// MI455X (gfx1250) — matrix operand order repaired and checked by reading, not compiled, not run on hardware
//
#include <hip/hip_runtime.h>
#include <cstdint>

// ---------------- problem constants ----------------
constexpr int cB = 32, cN = 1024, cF = 64;
constexpr int cP = 8, cn = 128, cm = 256, ce = 1024;
constexpr int cEF = 8, cT = 16, cH = 128, cW = 8, cG = 128, cR = 256, cL2 = 128;

// ---------------- WMMA types / helpers ----------------
typedef __attribute__((ext_vector_type(16))) __bf16 v16bf;
typedef __attribute__((ext_vector_type(8)))  __bf16 v8bf;
typedef __attribute__((ext_vector_type(4)))  __bf16 v4bf;
typedef __attribute__((ext_vector_type(8)))  float  v8f;
typedef __attribute__((ext_vector_type(4)))  float  vf4;

__device__ __forceinline__ __bf16 to_bf(float f) { return (__bf16)f; }

__device__ __forceinline__ v4bf cvt4(vf4 v) {
  v4bf r;
  r[0] = to_bf(v[0]); r[1] = to_bf(v[1]); r[2] = to_bf(v[2]); r[3] = to_bf(v[3]);
  return r;
}

__device__ __forceinline__ v8f zero8() {
  v8f z;
#pragma unroll
  for (int i = 0; i < 8; ++i) z[i] = 0.0f;
  return z;
}

__device__ __forceinline__ v8f wmma_bf16(v16bf a, v16bf b, v8f c) {
  // D = A(16x32) * B(32x16) + C, f32 accumulate
  return __builtin_amdgcn_wmma_f32_16x16x32_bf16(false, a, false, b, (short)0, c, false, false);
}

// A fragment 16x32 from LDS tile stored [M][K] row-major (leading dim ld), k offset k0.
// lane&15 = row M; half=0: slots0-7 -> K 0..7, 8-15 -> K 16..23; half=1: K 8..15 / 24..31.
// Contiguous 16B runs -> ds_load_b128 x2.
__device__ __forceinline__ v16bf frag_A(const __bf16* lds, int ld, int k0, int lane) {
  const int r = lane & 15;
  const int lh = lane >> 4;
  const __bf16* base = lds + r * ld + k0 + lh * 8;
  v8bf lo = *(const v8bf*)(base);       // K: lh*8 .. lh*8+7
  v8bf hi = *(const v8bf*)(base + 16);  // K: 16+lh*8 ..
  return __builtin_shufflevector(lo, hi, 0, 1, 2, 3, 4, 5, 6, 7, 8, 9, 10, 11, 12, 13, 14, 15);
}

// B fragment 32x16 where LDS holds B TRANSPOSED: [N][K] row-major, leading dim ld.
// B element (k, n) = lds[n*ld + k]; lane&15 = column N; half=0: K 0..7 / 16..23, half=1: K 8..15 / 24..31.
// Contiguous 16B runs -> ds_load_b128 x2.
__device__ __forceinline__ v16bf frag_BT(const __bf16* lds, int ld, int k0, int n0, int lane) {
  const __bf16* base = lds + (n0 + (lane & 15)) * ld + k0 + (lane >> 4) * 8;
  v8bf lo = *(const v8bf*)(base);
  v8bf hi = *(const v8bf*)(base + 16);
  return __builtin_shufflevector(lo, hi, 0, 1, 2, 3, 4, 5, 6, 7, 8, 9, 10, 11, 12, 13, 14, 15);
}

// ---------------- tiny precompute kernels (VALU) ----------------
__global__ void kbase_kernel(const float* __restrict__ ef, const float* __restrict__ k1w,
                             const float* __restrict__ k1b, float* __restrict__ kbase) {
  int idx = blockIdx.x * 256 + threadIdx.x;
  if (idx >= cP * ce * cH) return;
  int hh = idx % cH;
  int ee = (idx / cH) % ce;
  int p  = idx / (cH * ce);
  float acc = k1b[p * cH + hh];
  const float* efr = ef + (size_t)(p * ce + ee) * (cEF + 1);
  const float* wr  = k1w + (size_t)(p * cH + hh) * cEF;
#pragma unroll
  for (int c = 0; c < cEF; ++c) acc += wr[c] * efr[c];
  kbase[idx] = acc; // [p][e][h]
}

__global__ void bterm_kernel(const float* __restrict__ tin, const float* __restrict__ k2w,
                             float* __restrict__ bterm) {
  int idx = blockIdx.x * 256 + threadIdx.x;
  if (idx >= cB * cP * cH) return;
  int hh = idx % cH;
  int p  = (idx / cH) % cP;
  int b  = idx / (cH * cP);
  float acc = 0.0f;
#pragma unroll
  for (int t = 0; t < cT; ++t) acc += k2w[(p * cH + hh) * cT + t] * tin[b * cT + t];
  bterm[idx] = acc; // [b][p][h]
}

__global__ void hb_kernel(const float* __restrict__ tin, const float* __restrict__ cap,
                          const float* __restrict__ r1w, const float* __restrict__ r1b,
                          const float* __restrict__ r2w, float* __restrict__ hb) {
  int idx = blockIdx.x * 256 + threadIdx.x;
  if (idx >= cB * cP * cR) return;
  int r = idx % cR;
  int p = (idx / cR) % cP;
  int b = idx / (cR * cP);
  float acc = r1b[p * cR + r];
#pragma unroll
  for (int t = 0; t < cT; ++t) acc += r1w[(p * cR + r) * cT + t] * tin[b * cT + t];
  const float* r2r = r2w + (size_t)(p * cR + r) * cn;
  const float* cpr = cap + p * cn;
  for (int nn = 0; nn < cn; ++nn) acc += r2r[nn] * cpr[nn];
  hb[idx] = acc; // [b][p][r]
}

// ---------------- fused edge kernel ----------------
// Per (b,p, 128-row e-tile): rx = rinc@xn ; lx = linc^T@xg ;
// k = kbase + bterm + lx@k3^T + rx@k4^T ; kernel = relu(leaky(k,.02)@k5^T + k5b)
__global__ void fused_edge(const float* __restrict__ x, const float* __restrict__ linc,
                           const float* __restrict__ rinc,
                           const float* __restrict__ k3w, const float* __restrict__ k4w,
                           const float* __restrict__ k5w, const float* __restrict__ k5b,
                           const float* __restrict__ kbase, const float* __restrict__ bterm,
                           const int* __restrict__ nodes_idx, const int* __restrict__ nbr_idx,
                           float* __restrict__ rx_out, float* __restrict__ kern_out) {
  const int tid = threadIdx.x;
  const int lane = tid & 31, wave = tid >> 5;
  const int lh = lane >> 4, ln = lane & 15;
  const int bp = blockIdx.x;
  const int b = bp / cP, p = bp % cP;
  const int eBase = blockIdx.y * 128;

  extern __shared__ char smem_raw[];
  __bf16* sStageX = (__bf16*)smem_raw;   // [64 f][32 m]  (B^T layout)   (2048)
  __bf16* sStageW = sStageX + 2048;      // [128 e][32 k] (A layout)     (4096)
  __bf16* sK3     = sStageW + 4096;      // [128 h][64 f] (B^T)          (8192)
  __bf16* sK4     = sK3 + 8192;          // [128 h][64 f] (B^T)          (8192)
  __bf16* sK5     = sK4 + 8192;          // [16 wpad][128 h] (B^T)       (2048)
  __bf16* sRX     = sK5 + 2048;          // 8 waves x [16][64] (A)       (8192)
  __bf16* sLX     = sRX + 8192;          // 8 waves x [16][64] (A)       (8192)
  __bf16* sKt     = sLX + 8192;          // 8 waves x [16][128] (A)      (16384)

  // stage per-p weights once (vectorized: global_load_b128 + 8B LDS stores)
  for (int i = tid; i < cH * cF / 4; i += 256) {
    vf4 a3 = *(const vf4*)(k3w + (size_t)p * cH * cF + i * 4);
    vf4 a4 = *(const vf4*)(k4w + (size_t)p * cH * cF + i * 4);
    *(v4bf*)(sK3 + i * 4) = cvt4(a3);
    *(v4bf*)(sK4 + i * 4) = cvt4(a4);
  }
  for (int i = tid; i < 16 * cH / 4; i += 256) {
    int w = i >> 5;
    int h4 = (i & 31) * 4;
    v4bf pk;
    if (w < cW) {
      pk = cvt4(*(const vf4*)(k5w + (size_t)(p * cW + w) * cH + h4));
    } else {
      pk[0] = pk[1] = pk[2] = pk[3] = to_bf(0.0f);
    }
    *(v4bf*)(sK5 + i * 4) = pk;
  }

  // ---- phase 1: rx = rinc @ xn  (M=128 e-rows, K=m=256, N=F=64) ----
  v8f racc[4];
#pragma unroll
  for (int t = 0; t < 4; ++t) racc[t] = zero8();
  for (int mc = 0; mc < cm; mc += 32) {
    __syncthreads();
    // xn^T chunk: sStageX[f][m_local]; thread handles 4 m-rows at one f
    for (int i = tid; i < 512; i += 256) {
      int f = i & 63, rg = i >> 6; // rg 0..7
      v4bf pk;
#pragma unroll
      for (int j = 0; j < 4; ++j) {
        int node = nbr_idx[p * cm + mc + rg * 4 + j];
        pk[j] = to_bf(x[((size_t)b * cN + node) * cF + f]);
      }
      *(v4bf*)(sStageX + f * 32 + rg * 4) = pk;
    }
    // rinc chunk (A): [er][32k], float4 reads
    for (int i = tid; i < 1024; i += 256) {
      int er = i >> 3, kq = (i & 7) * 4;
      vf4 a = *(const vf4*)(rinc + ((size_t)(p * ce + eBase + er)) * cm + mc + kq);
      *(v4bf*)(sStageW + er * 32 + kq) = cvt4(a);
    }
    __syncthreads();
    v16bf a = frag_A(sStageW + wave * 16 * 32, 32, 0, lane);
#pragma unroll
    for (int ft = 0; ft < 4; ++ft)
      racc[ft] = wmma_bf16(a, frag_BT(sStageX, 32, 0, 16 * ft, lane), racc[ft]);
  }
  __syncthreads();
  // write rx tile to HBM and stash bf16 copy in LDS (A layout) for k4 pass
#pragma unroll
  for (int ft = 0; ft < 4; ++ft)
#pragma unroll
    for (int v = 0; v < 8; ++v) {
      int rloc = v + 8 * lh;
      int col = 16 * ft + ln;
      float val = racc[ft][v];
      int erow = eBase + 16 * wave + rloc;
      rx_out[(((size_t)(b * cP + p)) * ce + erow) * cF + col] = val;
      sRX[(wave * 16 + rloc) * cF + col] = to_bf(val);
    }

  // ---- phase 2: lx = linc^T @ xg  (K=n=128) ----
  v8f lacc[4];
#pragma unroll
  for (int t = 0; t < 4; ++t) lacc[t] = zero8();
  for (int nc = 0; nc < cn; nc += 32) {
    __syncthreads();
    for (int i = tid; i < 512; i += 256) {
      int f = i & 63, rg = i >> 6;
      v4bf pk;
#pragma unroll
      for (int j = 0; j < 4; ++j) {
        int node = nodes_idx[p * cn + nc + rg * 4 + j];
        pk[j] = to_bf(x[((size_t)b * cN + node) * cF + f]);
      }
      *(v4bf*)(sStageX + f * 32 + rg * 4) = pk;
    }
    // linc^T chunk (A): [er][k=n]; coalesced float4 over e, scatter bf16 stores
    for (int i = tid; i < 1024; i += 256) {
      int k = i & 31, eq = (i >> 5) * 4;
      vf4 a = *(const vf4*)(linc + ((size_t)(p * cn + nc + k)) * ce + eBase + eq);
#pragma unroll
      for (int j = 0; j < 4; ++j) sStageW[(eq + j) * 32 + k] = to_bf(a[j]);
    }
    __syncthreads();
    v16bf a = frag_A(sStageW + wave * 16 * 32, 32, 0, lane);
#pragma unroll
    for (int ft = 0; ft < 4; ++ft)
      lacc[ft] = wmma_bf16(a, frag_BT(sStageX, 32, 0, 16 * ft, lane), lacc[ft]);
  }
#pragma unroll
  for (int ft = 0; ft < 4; ++ft)
#pragma unroll
    for (int v = 0; v < 8; ++v)
      sLX[(wave * 16 + v + 8 * lh) * cF + 16 * ft + ln] = to_bf(lacc[ft][v]);
  __syncthreads();

  // ---- phase 3: k = kbase + bterm + lx@k3^T + rx@k4^T  ([16 e][128 H] per wave) ----
  v8f kacc[8];
#pragma unroll
  for (int ht = 0; ht < 8; ++ht)
#pragma unroll
    for (int v = 0; v < 8; ++v) {
      int erow = eBase + 16 * wave + v + 8 * lh;
      int hcol = 16 * ht + ln;
      kacc[ht][v] = kbase[((size_t)p * ce + erow) * cH + hcol] +
                    bterm[(b * cP + p) * cH + hcol];
    }
#pragma unroll
  for (int kc = 0; kc < cF; kc += 32) {
    v16bf a4 = frag_A(sRX + wave * 16 * cF, cF, kc, lane);
    v16bf a3 = frag_A(sLX + wave * 16 * cF, cF, kc, lane);
#pragma unroll
    for (int ht = 0; ht < 8; ++ht) {
      kacc[ht] = wmma_bf16(a3, frag_BT(sK3, cF, kc, 16 * ht, lane), kacc[ht]);
      kacc[ht] = wmma_bf16(a4, frag_BT(sK4, cF, kc, 16 * ht, lane), kacc[ht]);
    }
  }
  // leaky(0.02), round-trip through LDS as A operand
#pragma unroll
  for (int ht = 0; ht < 8; ++ht)
#pragma unroll
    for (int v = 0; v < 8; ++v) {
      float kv = kacc[ht][v];
      kv = (kv >= 0.0f) ? kv : 0.02f * kv;
      sKt[(wave * 16 + v + 8 * lh) * cH + 16 * ht + ln] = to_bf(kv);
    }
  __syncthreads();

  // ---- phase 4: kernel = relu(k @ k5^T + k5b)  (N padded 8->16) ----
  v8f wacc = zero8();
#pragma unroll
  for (int kc = 0; kc < cH; kc += 32)
    wacc = wmma_bf16(frag_A(sKt + wave * 16 * cH, cH, kc, lane),
                     frag_BT(sK5, cH, kc, 0, lane), wacc);
  if (ln < cW) {
#pragma unroll
    for (int v = 0; v < 8; ++v) {
      int erow = eBase + 16 * wave + v + 8 * lh;
      float val = wacc[v] + k5b[p * cW + ln];
      kern_out[(((size_t)(b * cP + p)) * ce + erow) * cW + ln] = fmaxf(val, 0.0f);
    }
  }
}

// ---------------- y = linc @ (kernel x rx)  (M=n=128, K=e=1024, N=W*F=512) ----------------
__global__ void y_kernel(const float* __restrict__ linc, const float* __restrict__ rx,
                         const float* __restrict__ kern, float* __restrict__ y) {
  const int tid = threadIdx.x;
  const int lane = tid & 31, wave = tid >> 5;
  const int lh = lane >> 4, ln = lane & 15;
  const int bp = blockIdx.x;
  const int b = bp / cP, p = bp % cP;
  const int colBase = blockIdx.y * 128;

  extern __shared__ char smem_raw[];
  __bf16* sZ = (__bf16*)smem_raw; // [128 col][32 e]  (B^T layout)
  __bf16* sA = sZ + 128 * 32;     // [128 n][32 e]    (A layout)

  v8f acc[8];
#pragma unroll
  for (int t = 0; t < 8; ++t) acc[t] = zero8();

  const size_t base_e = ((size_t)(b * cP + p)) * ce;
  for (int ec = 0; ec < ce; ec += 32) {
    __syncthreads();
    // z^T chunk: sZ[col][e_local] = kern[e,w]*rx[e,f]; float4 over f, scatter stores
    for (int i = tid; i < 1024; i += 256) {
      int er = i >> 5, cq = (i & 31) * 4;
      size_t eidx = base_e + ec + er;
      int cg0 = colBase + cq;
      int w = cg0 >> 6, f0 = cg0 & 63; // 4-col quad never crosses a w boundary
      float kv = kern[eidx * cW + w];
      vf4 rv = *(const vf4*)(rx + eidx * cF + f0);
#pragma unroll
      for (int j = 0; j < 4; ++j) sZ[(cq + j) * 32 + er] = to_bf(kv * rv[j]);
    }
    // linc chunk (A): [n][32 e], float4 reads
    for (int i = tid; i < 1024; i += 256) {
      int nr = i >> 3, kq = (i & 7) * 4;
      vf4 a = *(const vf4*)(linc + ((size_t)(p * cn + nr)) * ce + ec + kq);
      *(v4bf*)(sA + nr * 32 + kq) = cvt4(a);
    }
    __syncthreads();
    v16bf a = frag_A(sA + wave * 16 * 32, 32, 0, lane);
#pragma unroll
    for (int nt = 0; nt < 8; ++nt)
      acc[nt] = wmma_bf16(a, frag_BT(sZ, 32, 0, 16 * nt, lane), acc[nt]);
  }
#pragma unroll
  for (int nt = 0; nt < 8; ++nt)
#pragma unroll
    for (int v = 0; v < 8; ++v) {
      int nrow = 16 * wave + v + 8 * lh;
      int col = colBase + 16 * nt + ln;
      y[(((size_t)(b * cP + p)) * cn + nrow) * (cW * cF) + col] = acc[nt][v];
    }
}

// ---------------- y2 = relu(y @ gc^T + gcb)  (M=128, K=512, N=G=128) ----------------
__global__ void gc_kernel(const float* __restrict__ y, const float* __restrict__ gcw,
                          const float* __restrict__ gcb, float* __restrict__ y2) {
  const int tid = threadIdx.x;
  const int lane = tid & 31, wave = tid >> 5;
  const int lh = lane >> 4, ln = lane & 15;
  const int bp = blockIdx.x;
  const int b = bp / cP, p = bp % cP;

  extern __shared__ char smem_raw[];
  __bf16* sA = (__bf16*)smem_raw; // [128 n][32 k] (A layout)
  __bf16* sB = sA + 128 * 32;     // [128 g][32 k] (B^T layout, gcw native)

  v8f acc[8];
#pragma unroll
  for (int t = 0; t < 8; ++t) acc[t] = zero8();

  const size_t ybase = ((size_t)(b * cP + p)) * cn * (cW * cF);
  for (int kc = 0; kc < cW * cF; kc += 32) {
    __syncthreads();
    for (int i = tid; i < 1024; i += 256) {
      int nr = i >> 3, kq = (i & 7) * 4;
      vf4 a = *(const vf4*)(y + ybase + (size_t)nr * (cW * cF) + kc + kq);
      *(v4bf*)(sA + nr * 32 + kq) = cvt4(a);
    }
    for (int i = tid; i < 1024; i += 256) {
      int g = i >> 3, kq = (i & 7) * 4;
      vf4 a = *(const vf4*)(gcw + ((size_t)(p * cG + g)) * (cW * cF) + kc + kq);
      *(v4bf*)(sB + g * 32 + kq) = cvt4(a);
    }
    __syncthreads();
    v16bf a = frag_A(sA + wave * 16 * 32, 32, 0, lane);
#pragma unroll
    for (int gt = 0; gt < 8; ++gt)
      acc[gt] = wmma_bf16(a, frag_BT(sB, 32, 0, 16 * gt, lane), acc[gt]);
  }
#pragma unroll
  for (int gt = 0; gt < 8; ++gt)
#pragma unroll
    for (int v = 0; v < 8; ++v) {
      int nrow = 16 * wave + v + 8 * lh;
      int g = 16 * gt + ln;
      float val = acc[gt][v] + gcb[p * cG + g];
      y2[(((size_t)(b * cP + p)) * cn + nrow) * cG + g] = fmaxf(val, 0.0f);
    }
}

// ---------------- h = y2flat @ r0^T + hb  (per p: M=B=32, K=16384, N=R=256) ----------------
// r0_w is 134MB (the dominant HBM stream) -> double-buffered async DMA to LDS
// (global_load_async_to_lds_b128, ASYNCcnt), convert+WMMA overlapped with next tile's DMA.
__global__ void h_kernel(const float* __restrict__ y2, const float* __restrict__ r0w,
                         const float* __restrict__ hb, float* __restrict__ h) {
  const int tid = threadIdx.x;
  const int lane = tid & 31, wave = tid >> 5;
  const int lh = lane >> 4, ln = lane & 15;
  const int p = blockIdx.x;
  const int rBase = blockIdx.y * 128;
  const int K = cn * cG; // 16384

  extern __shared__ char smem_raw[];
  __bf16* sA  = (__bf16*)smem_raw;        // [32 b][32 k]  (A layout)           2048 B
  __bf16* sB  = sA + 32 * 32;             // [128 r][32 k] (B^T layout, bf16)   8192 B
  float*  sBf = (float*)(sB + 128 * 32);  // 2 x [128 r][32 k] f32 DMA buffers  32768 B

  v8f acc[2];
  acc[0] = zero8();
  acc[1] = zero8();

  const uint64_t rb = (uint64_t)(r0w + ((size_t)(p * cR + rBase)) * K);

  // each thread issues 4 x b128 async copies covering the [128 r][32 k] f32 tile
  auto issue_chunk = [&](int kc, float* buf) {
#pragma unroll
    for (int j = 0; j < 4; ++j) {
      int idx = tid + 256 * j;          // 0..1023
      int r = idx >> 3, kq = (idx & 7) * 4;
      unsigned voff = (unsigned)(r * K + kc + kq) * 4u; // byte offset, < 16 MB
      unsigned laddr = (unsigned)(size_t)(void*)(buf + r * 32 + kq);
      asm volatile("global_load_async_to_lds_b128 %0, %1, %2"
                   :: "v"(laddr), "v"(voff), "s"(rb)
                   : "memory");
    }
  };

  issue_chunk(0, sBf);
  for (int c = 0; c < K / 32; ++c) {
    const int kc = c * 32;
    float* cur = sBf + (c & 1) * (128 * 32);
    float* nxt = sBf + ((c + 1) & 1) * (128 * 32);
    asm volatile("s_wait_asynccnt 0x0" ::: "memory"); // own wave's DMA landed
    __syncthreads();                                  // => all waves' DMA landed
    if (c + 1 < K / 32) issue_chunk(kc + 32, nxt);    // overlap next tile's DMA
    // convert f32 tile -> bf16 B^T tile
    for (int i = tid; i < 1024; i += 256) {
      vf4 a = *(const vf4*)(cur + i * 4);
      *(v4bf*)(sB + i * 4) = cvt4(a);
    }
    // stage A (y2 rows, M=32)
    {
      int i = tid;
      if (i < 256) {
        int br = i >> 3, kq = (i & 7) * 4;
        vf4 a = *(const vf4*)(y2 + ((size_t)(br * cP + p)) * K + kc + kq);
        *(v4bf*)(sA + br * 32 + kq) = cvt4(a);
      }
    }
    __syncthreads();
    v16bf bfr = frag_BT(sB, 32, 0, 16 * wave, lane);
#pragma unroll
    for (int mt = 0; mt < 2; ++mt)
      acc[mt] = wmma_bf16(frag_A(sA + mt * 16 * 32, 32, 0, lane), bfr, acc[mt]);
  }
#pragma unroll
  for (int mt = 0; mt < 2; ++mt)
#pragma unroll
    for (int v = 0; v < 8; ++v) {
      int brow = 16 * mt + v + 8 * lh;
      int rcol = rBase + 16 * wave + ln;
      size_t idx = ((size_t)(brow * cP + p)) * cR + rcol;
      h[idx] = acc[mt][v] + hb[idx];
    }
}

// ---------------- tail (small VALU kernels) ----------------
__global__ void wv_kernel(const float* __restrict__ h, const float* __restrict__ r3w,
                          const float* __restrict__ r3b, float* __restrict__ wv) {
  int idx = blockIdx.x * 256 + threadIdx.x;
  if (idx >= cB * cP * cn) return;
  int nn = idx % cn;
  int p = (idx / cn) % cP;
  int b = idx / (cn * cP);
  float acc = r3b[p * cn + nn];
  const float* hr = h + (size_t)(b * cP + p) * cR;
  const float* wr = r3w + (size_t)(p * cn + nn) * cR;
  for (int r = 0; r < cR; ++r) {
    float hv = hr[r];
    hv = (hv >= 0.0f) ? hv : 0.01f * hv;
    acc += wr[r] * hv;
  }
  wv[idx] = fmaxf(acc, 0.0f);
}

__global__ void res_kernel(const float* __restrict__ wv, const float* __restrict__ y2,
                           float* __restrict__ res) {
  int idx = blockIdx.x * 256 + threadIdx.x;
  if (idx >= cB * cP * cG) return;
  int g = idx % cG;
  int p = (idx / cG) % cP;
  int b = idx / (cG * cP);
  const float* w = wv + (size_t)(b * cP + p) * cn;
  const float* yr = y2 + (size_t)(b * cP + p) * cn * cG + g;
  float acc = 0.0f;
  for (int nn = 0; nn < cn; ++nn) acc += w[nn] * yr[(size_t)nn * cG];
  res[idx] = acc;
}

__global__ void out_kernel(const float* __restrict__ res, const float* __restrict__ convw,
                           const float* __restrict__ convb, float* __restrict__ out) {
  int idx = blockIdx.x * 256 + threadIdx.x;
  if (idx >= cB * cL2) return;
  int l = idx % cL2;
  int b = idx / cL2;
  float acc = convb[l];
  for (int g = 0; g < cG; ++g) {
    const float* cw = convw + ((size_t)l * cG + g) * cP;
    const float* rr = res + (size_t)b * cP * cG + g;
#pragma unroll
    for (int p = 0; p < cP; ++p) acc += cw[p] * rr[(size_t)p * cG];
  }
  out[idx] = (acc >= 0.0f) ? acc : 0.02f * acc;
}

// ---------------- host launcher ----------------
extern "C" void kernel_launch(void* const* d_in, const int* in_sizes, int n_in,
                              void* d_out, int out_size, void* d_ws, size_t ws_size,
                              hipStream_t stream) {
  (void)in_sizes; (void)n_in; (void)out_size; (void)ws_size;
  const float* x    = (const float*)d_in[0];
  const float* tin  = (const float*)d_in[1];
  const float* ef   = (const float*)d_in[2];
  const float* linc = (const float*)d_in[3];
  const float* rinc = (const float*)d_in[4];
  const float* cap  = (const float*)d_in[5];
  const float* k1w  = (const float*)d_in[6];
  const float* k1b  = (const float*)d_in[7];
  const float* k2w  = (const float*)d_in[8];
  const float* k3w  = (const float*)d_in[9];
  const float* k4w  = (const float*)d_in[10];
  const float* k5w  = (const float*)d_in[11];
  const float* k5b  = (const float*)d_in[12];
  const float* gcw  = (const float*)d_in[13];
  const float* gcb  = (const float*)d_in[14];
  const float* r0w  = (const float*)d_in[15];
  const float* r1w  = (const float*)d_in[16];
  const float* r1b  = (const float*)d_in[17];
  const float* r2w  = (const float*)d_in[18];
  const float* r3w  = (const float*)d_in[19];
  const float* r3b  = (const float*)d_in[20];
  const float* convw = (const float*)d_in[21];
  const float* convb = (const float*)d_in[22];
  const int* nodes_idx = (const int*)d_in[23];
  const int* nbr_idx   = (const int*)d_in[24];
  float* out = (float*)d_out;
  float* ws  = (float*)d_ws;

  // ---- workspace carve (floats); y2 aliases rx (rx dead after y_kernel) ----
  float* rx    = ws;                                     // B*P*e*F   = 16,777,216
  float* kern  = rx    + (size_t)cB * cP * ce * cF;      // B*P*e*W   =  2,097,152
  float* kbase = kern  + (size_t)cB * cP * ce * cW;      // P*e*H     =  1,048,576
  float* bterm = kbase + (size_t)cP * ce * cH;           // B*P*H     =     32,768
  float* ybuf  = bterm + (size_t)cB * cP * cH;           // B*P*n*W*F = 16,777,216
  float* hbb   = ybuf  + (size_t)cB * cP * cn * cW * cF; // B*P*R     =     65,536
  float* hbuf  = hbb   + (size_t)cB * cP * cR;
  float* wv    = hbuf  + (size_t)cB * cP * cR;           // B*P*n
  float* resb  = wv    + (size_t)cB * cP * cn;           // B*P*G
  float* y2    = rx;                                     // alias

  constexpr size_t FUSED_LDS = (size_t)(2048 + 4096 + 8192 + 8192 + 2048 + 8192 + 8192 + 16384) * 2;
  constexpr size_t Y_LDS  = (size_t)(128 * 32 + 128 * 32) * 2;
  constexpr size_t GC_LDS = (size_t)(128 * 32 + 128 * 32) * 2;
  constexpr size_t H_LDS  = (size_t)(32 * 32 + 128 * 32) * 2 + (size_t)2 * 128 * 32 * 4;

  kbase_kernel<<<(cP * ce * cH + 255) / 256, 256, 0, stream>>>(ef, k1w, k1b, kbase);
  bterm_kernel<<<(cB * cP * cH + 255) / 256, 256, 0, stream>>>(tin, k2w, bterm);
  hb_kernel<<<(cB * cP * cR + 255) / 256, 256, 0, stream>>>(tin, cap, r1w, r1b, r2w, hbb);

  fused_edge<<<dim3(cB * cP, ce / 128), 256, FUSED_LDS, stream>>>(
      x, linc, rinc, k3w, k4w, k5w, k5b, kbase, bterm, nodes_idx, nbr_idx, rx, kern);

  y_kernel<<<dim3(cB * cP, 4), 256, Y_LDS, stream>>>(linc, rx, kern, ybuf);
  gc_kernel<<<dim3(cB * cP), 256, GC_LDS, stream>>>(ybuf, gcw, gcb, y2);
  h_kernel<<<dim3(cP, 2), 256, H_LDS, stream>>>(y2, r0w, hbb, hbuf);

  wv_kernel<<<(cB * cP * cn + 255) / 256, 256, 0, stream>>>(hbuf, r3w, r3b, wv);
  res_kernel<<<(cB * cP * cG + 255) / 256, 256, 0, stream>>>(wv, y2, resb);
  out_kernel<<<(cB * cL2 + 255) / 256, 256, 0, stream>>>(resb, convw, convb, out);
}
